// TorchMLPFloat8Reference_55087250538623
// MI455X (gfx1250) — matrix operand order repaired and checked by reading, not compiled, not run on hardware
//
#include <hip/hip_runtime.h>
#include <hip/hip_bf16.h>
#include <stdint.h>

// Problem dims (compile-time constants from the reference).
#define HIDDEN 4096
#define FFL    14336
#define BATCH  2048

// Tiling: block = 128(M) x 64(N), K-tile 64. 256 threads = 8 wave32 waves,
// wave grid 4(M) x 2(N); each wave owns a 32x32 output patch = 2x2 WMMA frags.
#define MT 128
#define NT 64
#define KT 64
#define ASTR (KT + 8)   // LDS row stride in bf16 elems (144B rows, 16B aligned)
#define WSTR (KT + 8)

typedef __bf16 bf16;
typedef bf16  bf16x16 __attribute__((ext_vector_type(16)));
typedef bf16  bf16x8  __attribute__((ext_vector_type(8)));
typedef bf16  bf16x4  __attribute__((ext_vector_type(4)));
typedef float f32x8   __attribute__((ext_vector_type(8)));

union V16 { bf16x16 v; bf16x8 h[2]; };

// ---- CDNA5 async global->LDS copy (ASYNCcnt path, ISA §15.18.3 op 98). ----
// Each active lane copies 16 bytes from its global address to its LDS address.
// Generic LDS pointers carry the LDS byte offset in their low 32 bits
// (ISA §10.2 aperture mapping), so truncation yields the VDST LDS address.
__device__ __forceinline__ void async_copy_b128(const void* lds_ptr, const void* gptr) {
  uint32_t lds_off = (uint32_t)(uintptr_t)lds_ptr;
  uint64_t gaddr   = (uint64_t)(uintptr_t)gptr;
  asm volatile("global_load_async_to_lds_b128 %0, %1, off"
               :: "v"(lds_off), "v"(gaddr)
               : "memory");
}
__device__ __forceinline__ void wait_async0() {
  asm volatile("s_wait_asynccnt 0x0" ::: "memory");
}

// ---- WMMA fragment loaders per ISA §7.12.2 (wave32, 16-bit operands). ----
// A (16x32 MxK): lane r=lane&15 holds row M=r; lanes 0-15 carry K=[kk..kk+7]
// then K=[kk+16..kk+23]; lanes 16-31 carry K=[kk+8..kk+15],[kk+24..kk+31].
__device__ __forceinline__ bf16x16 load_a_frag(const bf16* As, int mrow, int kk, int lane) {
  const int r = lane & 15, half = lane >> 4;
  const bf16* p = As + (size_t)(mrow + r) * ASTR + kk + half * 8;
  V16 a;
  a.h[0] = *(const bf16x8*)p;
  a.h[1] = *(const bf16x8*)(p + 16);
  return a.v;
}
// B (32x16 KxN): lane n=lane&15 holds column N=n; lanes 0-15 carry K=[kk..kk+7]
// then K=[kk+16..kk+23]; lanes 16-31 carry K=[kk+8..kk+15],[kk+24..kk+31] (same as A).
__device__ __forceinline__ bf16x16 load_b_frag(const bf16* Ws, int ncol, int kk, int lane) {
  const int n = lane & 15, half = lane >> 4;
  const bf16* p = Ws + (size_t)(ncol + n) * WSTR + kk + half * 8;
  V16 b;
  b.h[0] = *(const bf16x8*)p;
  b.h[1] = *(const bf16x8*)(p + 16);
  return b.v;
}

// Dequantize 4 f32 weights by row scale, store packed bf16x4 (8B) into LDS.
__device__ __forceinline__ void dequant_store4(bf16* dst, const float4 w, float s) {
  bf16x4 t;
  t.x = (bf16)(w.x * s); t.y = (bf16)(w.y * s);
  t.z = (bf16)(w.z * s); t.w = (bf16)(w.w * s);
  *(bf16x4*)dst = t;
}

__device__ __forceinline__ f32x8 wmma_bf16(bf16x16 a, bf16x16 b, f32x8 c) {
  return __builtin_amdgcn_wmma_f32_16x16x32_bf16(
      /*neg_a=*/false, a, /*neg_b=*/false, b,
      /*c_mod=*/(short)0, c, /*reuse_a=*/false, /*reuse_b=*/false);
}

// ============================================================================
// Kernel 1: hidden[b,f] = silu(x @ gate_w^T) * (x @ up_w^T), dequant fused.
// ============================================================================
__global__ __launch_bounds__(256, 1) void mlp_gate_up_silu(
    const bf16* __restrict__ x,
    const float* __restrict__ gw, const float* __restrict__ gsc,
    const float* __restrict__ uw, const float* __restrict__ usc,
    bf16* __restrict__ hidden)
{
  __shared__ __align__(16) bf16 As[MT * ASTR];
  __shared__ __align__(16) bf16 Gs[NT * WSTR];
  __shared__ __align__(16) bf16 Us[NT * WSTR];

  const int tid  = threadIdx.x;
  const int lane = tid & 31;
  const int wave = tid >> 5;
  const int wm   = wave & 3;   // 4 waves along M, 32 rows each
  const int wn   = wave >> 2;  // 2 waves along N, 32 cols each
  const int m0   = blockIdx.y * MT;
  const int f0   = blockIdx.x * NT;

  const f32x8 fz = {0.f,0.f,0.f,0.f,0.f,0.f,0.f,0.f};
  f32x8 accg[2][2], accu[2][2];
  #pragma unroll
  for (int i = 0; i < 2; ++i)
    #pragma unroll
    for (int j = 0; j < 2; ++j) { accg[i][j] = fz; accu[i][j] = fz; }

  for (int k0 = 0; k0 < HIDDEN; k0 += KT) {
    // Async-copy x tile (128 x 64 bf16 = 16KB): 1024 16B chunks, 4 per thread.
    #pragma unroll
    for (int i = 0; i < 4; ++i) {
      const int ch = tid + i * 256;
      const int row = ch >> 3, c8 = ch & 7;
      async_copy_b128(&As[row * ASTR + c8 * 8],
                      x + (size_t)(m0 + row) * HIDDEN + k0 + c8 * 8);
    }
    // Dequantize gate/up weight tiles (64 x 64 f32 each) into bf16 LDS tiles.
    #pragma unroll
    for (int i = 0; i < 4; ++i) {
      const int ch = tid + i * 256;
      const int row = ch >> 4, c4 = ch & 15;
      const size_t goff = (size_t)(f0 + row) * HIDDEN + k0 + c4 * 4;
      dequant_store4(&Gs[row * WSTR + c4 * 4], *(const float4*)(gw + goff), gsc[f0 + row]);
      dequant_store4(&Us[row * WSTR + c4 * 4], *(const float4*)(uw + goff), usc[f0 + row]);
    }
    wait_async0();
    __syncthreads();

    #pragma unroll
    for (int kk = 0; kk < KT; kk += 32) {
      bf16x16 a0 = load_a_frag(As, wm * 32,      kk, lane);
      bf16x16 a1 = load_a_frag(As, wm * 32 + 16, kk, lane);
      bf16x16 bg0 = load_b_frag(Gs, wn * 32,      kk, lane);
      bf16x16 bg1 = load_b_frag(Gs, wn * 32 + 16, kk, lane);
      bf16x16 bu0 = load_b_frag(Us, wn * 32,      kk, lane);
      bf16x16 bu1 = load_b_frag(Us, wn * 32 + 16, kk, lane);
      accg[0][0] = wmma_bf16(a0, bg0, accg[0][0]);
      accg[0][1] = wmma_bf16(a0, bg1, accg[0][1]);
      accg[1][0] = wmma_bf16(a1, bg0, accg[1][0]);
      accg[1][1] = wmma_bf16(a1, bg1, accg[1][1]);
      accu[0][0] = wmma_bf16(a0, bu0, accu[0][0]);
      accu[0][1] = wmma_bf16(a0, bu1, accu[0][1]);
      accu[1][0] = wmma_bf16(a1, bu0, accu[1][0]);
      accu[1][1] = wmma_bf16(a1, bu1, accu[1][1]);
    }
    __syncthreads();
  }

  // Epilogue: C/D layout (ISA §7.12.2): VGPR v, lane l -> M = v + 8*(l>>4), N = l&15.
  const int half = lane >> 4, nn = lane & 15;
  #pragma unroll
  for (int i = 0; i < 2; ++i)
    #pragma unroll
    for (int j = 0; j < 2; ++j) {
      const int col   = f0 + wn * 32 + j * 16 + nn;
      const int rbase = m0 + wm * 32 + i * 16 + half * 8;
      #pragma unroll
      for (int v = 0; v < 8; ++v) {
        const float g = accg[i][j][v];
        const float u = accu[i][j][v];
        const float hval = (g / (1.f + __expf(-g))) * u;  // silu(g) * u
        hidden[(size_t)(rbase + v) * FFL + col] = (bf16)hval;
      }
    }
}

// ============================================================================
// Kernel 2: out[b,h] = hidden @ down_w^T, dequant fused.
// ============================================================================
__global__ __launch_bounds__(256, 1) void mlp_down(
    const bf16* __restrict__ hidden,
    const float* __restrict__ dw, const float* __restrict__ dsc,
    bf16* __restrict__ out)
{
  __shared__ __align__(16) bf16 As[MT * ASTR];
  __shared__ __align__(16) bf16 Ws[NT * WSTR];

  const int tid  = threadIdx.x;
  const int lane = tid & 31;
  const int wave = tid >> 5;
  const int wm   = wave & 3;
  const int wn   = wave >> 2;
  const int m0   = blockIdx.y * MT;
  const int h0   = blockIdx.x * NT;

  const f32x8 fz = {0.f,0.f,0.f,0.f,0.f,0.f,0.f,0.f};
  f32x8 acc[2][2];
  #pragma unroll
  for (int i = 0; i < 2; ++i)
    #pragma unroll
    for (int j = 0; j < 2; ++j) acc[i][j] = fz;

  for (int k0 = 0; k0 < FFL; k0 += KT) {
    #pragma unroll
    for (int i = 0; i < 4; ++i) {
      const int ch = tid + i * 256;
      const int row = ch >> 3, c8 = ch & 7;
      async_copy_b128(&As[row * ASTR + c8 * 8],
                      hidden + (size_t)(m0 + row) * FFL + k0 + c8 * 8);
    }
    #pragma unroll
    for (int i = 0; i < 4; ++i) {
      const int ch = tid + i * 256;
      const int row = ch >> 4, c4 = ch & 15;
      const size_t goff = (size_t)(h0 + row) * FFL + k0 + c4 * 4;
      dequant_store4(&Ws[row * WSTR + c4 * 4], *(const float4*)(dw + goff), dsc[h0 + row]);
    }
    wait_async0();
    __syncthreads();

    #pragma unroll
    for (int kk = 0; kk < KT; kk += 32) {
      bf16x16 a0 = load_a_frag(As, wm * 32,      kk, lane);
      bf16x16 a1 = load_a_frag(As, wm * 32 + 16, kk, lane);
      bf16x16 b0 = load_b_frag(Ws, wn * 32,      kk, lane);
      bf16x16 b1 = load_b_frag(Ws, wn * 32 + 16, kk, lane);
      acc[0][0] = wmma_bf16(a0, b0, acc[0][0]);
      acc[0][1] = wmma_bf16(a0, b1, acc[0][1]);
      acc[1][0] = wmma_bf16(a1, b0, acc[1][0]);
      acc[1][1] = wmma_bf16(a1, b1, acc[1][1]);
    }
    __syncthreads();
  }

  const int half = lane >> 4, nn = lane & 15;
  #pragma unroll
  for (int i = 0; i < 2; ++i)
    #pragma unroll
    for (int j = 0; j < 2; ++j) {
      const int col   = h0 + wn * 32 + j * 16 + nn;
      const int rbase = m0 + wm * 32 + i * 16 + half * 8;
      #pragma unroll
      for (int v = 0; v < 8; ++v)
        out[(size_t)(rbase + v) * HIDDEN + col] = (bf16)acc[i][j][v];
    }
}

// ============================================================================
extern "C" void kernel_launch(void* const* d_in, const int* in_sizes, int n_in,
                              void* d_out, int out_size, void* d_ws, size_t ws_size,
                              hipStream_t stream) {
  (void)in_sizes; (void)n_in; (void)out_size; (void)ws_size;
  const bf16*  x   = (const bf16*)d_in[0];
  const float* gw  = (const float*)d_in[1];
  const float* uw  = (const float*)d_in[2];
  const float* dw  = (const float*)d_in[3];
  const float* gsc = (const float*)d_in[4];
  const float* usc = (const float*)d_in[5];
  const float* dsc = (const float*)d_in[6];
  bf16* outp   = (bf16*)d_out;
  bf16* hidden = (bf16*)d_ws;   // 2048 x 14336 bf16 ~= 58.7 MB scratch

  dim3 blk(256);
  dim3 g1(FFL / NT, BATCH / MT);     // 224 x 16
  mlp_gate_up_silu<<<g1, blk, 0, stream>>>(x, gw, gsc, uw, usc, hidden);

  dim3 g2(HIDDEN / NT, BATCH / MT);  // 64 x 16
  mlp_down<<<g2, blk, 0, stream>>>(hidden, dw, dsc, outp);
}
